// seqFusionAttentionNoBias_48782238548117
// MI455X (gfx1250) — matrix operand order repaired and checked by reading, not compiled, not run on hardware
//
#include <hip/hip_runtime.h>
#include <hip/hip_bf16.h>

typedef __attribute__((ext_vector_type(16))) _Float16 v16h;
typedef __attribute__((ext_vector_type(8)))  _Float16 v8h;
typedef __attribute__((ext_vector_type(4)))  _Float16 v4h;
typedef __attribute__((ext_vector_type(8)))  float    v8f;

#define H_    16
#define C_    64
#define D_    1024
#define NB    2
#define SEQ   1024
#define MROWS (NB * SEQ)   // 2048

// ---------------------------------------------------------------------------
// fp32 -> f16 conversion, 4 elements/thread (b128 in, b64 out)
// ---------------------------------------------------------------------------
__global__ void cvt_f32_to_f16(const float* __restrict__ src,
                               _Float16* __restrict__ dst, int n) {
    int i = (blockIdx.x * blockDim.x + threadIdx.x) * 4;
    if (i < n) {
        float4 f = *(const float4*)(src + i);
        v4h h;
        h.x = (_Float16)f.x; h.y = (_Float16)f.y;
        h.z = (_Float16)f.z; h.w = (_Float16)f.w;
        *(v4h*)(dst + i) = h;
    }
}

// ---------------------------------------------------------------------------
// Transpose + convert: in fp32 [K][N] -> out f16 [N][K]  (LDS 32x32 tile)
// grid (N/32, K/32), block (32,8)
// ---------------------------------------------------------------------------
__global__ void transpose_cvt(const float* __restrict__ in,
                              _Float16* __restrict__ out, int K, int N) {
    __shared__ float tile[32][33];
    const int n0 = blockIdx.x * 32;
    const int k0 = blockIdx.y * 32;
    const int tx = threadIdx.x;
#pragma unroll
    for (int i = threadIdx.y; i < 32; i += 8)
        tile[i][tx] = in[(size_t)(k0 + i) * N + n0 + tx];
    __syncthreads();
#pragma unroll
    for (int i = threadIdx.y; i < 32; i += 8)
        out[(size_t)(n0 + i) * K + k0 + tx] = (_Float16)tile[tx][i];
}

// ---------------------------------------------------------------------------
// Per-head V transpose: in f16 [nb*SEQ + t][h*C + c] -> out f16 [nb][h*C + c][t]
// grid (SEQ/32, D/32, NB), block (32,8)
// ---------------------------------------------------------------------------
__global__ void transpose_V(const _Float16* __restrict__ in,
                            _Float16* __restrict__ out) {
    __shared__ _Float16 tile[32][34];
    const int t0 = blockIdx.x * 32;
    const int c0 = blockIdx.y * 32;
    const int nb = blockIdx.z;
    const int tx = threadIdx.x;
#pragma unroll
    for (int i = threadIdx.y; i < 32; i += 8)
        tile[i][tx] = in[(size_t)(nb * SEQ + t0 + i) * D_ + c0 + tx];
    __syncthreads();
#pragma unroll
    for (int i = threadIdx.y; i < 32; i += 8)
        out[(size_t)(nb * D_ + c0 + i) * SEQ + t0 + tx] = tile[tx][i];
}

// ---------------------------------------------------------------------------
// WMMA GEMM with PRE-TRANSPOSED B:
//   C = act(alpha * A(f16, MxK row-major) @ Bt(f16, NxK row-major)^T + bias)
// block = 128 threads = 4 waves, block tile 128x128, wave tile 64x64.
// act: 0 = none, 1 = sigmoid.  Cf (fp32) and/or Ch (f16) outputs, nullable.
// Every WMMA operand is two 16B loads per lane, starting 32B apart.
// ---------------------------------------------------------------------------
__global__ __launch_bounds__(128)
void wmma_gemm(const _Float16* __restrict__ A, const _Float16* __restrict__ Bt,
               const float* __restrict__ bias, float alpha, int act,
               float* __restrict__ Cf, _Float16* __restrict__ Ch,
               int M, int Ncols, int K)
{
    const int lane = threadIdx.x & 31;
    const int wave = threadIdx.x >> 5;
    const int col  = lane & 15;
    const int hi   = lane >> 4;

    const int mbase = blockIdx.x * 128 + (wave >> 1) * 64;
    const int nbase = blockIdx.y * 128 + (wave & 1) * 64;

    v8f acc[4][4];
#pragma unroll
    for (int a = 0; a < 4; a++)
#pragma unroll
        for (int b = 0; b < 4; b++) acc[a][b] = 0.0f;

    for (int kk = 0; kk < K; kk += 32) {
        // A fragment: lane row, K = {kb..kb+7, kb+16..kb+23}, kb = 8*hi
        v16h afrag[4];
#pragma unroll
        for (int a = 0; a < 4; a++) {
            const _Float16* p = A + (size_t)(mbase + a * 16 + col) * K + kk + hi * 8;
            union { v16h v; v8h h[2]; } u;
            u.h[0] = *(const v8h*)p;
            u.h[1] = *(const v8h*)(p + 16);
            afrag[a] = u.v;
        }
        // B fragment from Bt: lane col n, K = {kb..kb+7, kb+16..kb+23}, kb = 8*hi
        v16h bfrag[4];
#pragma unroll
        for (int b = 0; b < 4; b++) {
            const _Float16* q = Bt + (size_t)(nbase + b * 16 + col) * K + kk + hi * 8;
            union { v16h v; v8h h[2]; } w;
            w.h[0] = *(const v8h*)q;
            w.h[1] = *(const v8h*)(q + 16);
            bfrag[b] = w.v;
        }
#pragma unroll
        for (int b = 0; b < 4; b++)
#pragma unroll
            for (int a = 0; a < 4; a++)
                acc[a][b] = __builtin_amdgcn_wmma_f32_16x16x32_f16(
                    false, afrag[a], false, bfrag[b], (short)0, acc[a][b], false, false);
    }

#pragma unroll
    for (int a = 0; a < 4; a++) {
        const int m0 = mbase + a * 16 + hi * 8;
#pragma unroll
        for (int b = 0; b < 4; b++) {
            const int n0 = nbase + b * 16 + col;
#pragma unroll
            for (int r = 0; r < 8; r++) {
                float v = acc[a][b][r] * alpha;
                if (bias) v += bias[n0];
                if (act == 1) v = 1.0f / (1.0f + __expf(-v));
                const size_t idx = (size_t)(m0 + r) * Ncols + n0;
                if (Cf) Cf[idx] = v;
                if (Ch) Ch[idx] = (_Float16)v;
            }
        }
    }
}

// ---------------------------------------------------------------------------
// Flash-style gated attention. One wave per (n, h, 16-row s-tile).
// Q is pre-scaled by 1/sqrt(C). Online softmax; P staged via LDS to build
// the WMMA A-fragment; V pre-transposed so the P@V B-frag is contiguous.
// ---------------------------------------------------------------------------
__global__ __launch_bounds__(32)
void attn_kernel(const _Float16* __restrict__ Q,
                 const _Float16* __restrict__ Kp,
                 const _Float16* __restrict__ Vt,    // [nb][h*C + c][SEQ]
                 const float*    __restrict__ G,
                 const unsigned char* __restrict__ seqMask,
                 const float*    __restrict__ attMask,   // [SEQ, SEQ]
                 _Float16* __restrict__ O)
{
    __shared__ _Float16 lds[16 * 32];

    const int lane  = threadIdx.x;
    const int col   = lane & 15;
    const int hi    = lane >> 4;
    const int stile = blockIdx.x;
    const int h     = blockIdx.y;
    const int nb    = blockIdx.z;
    const int sbase = stile * 16;
    const size_t rowOff = (size_t)nb * SEQ;

    // Q fragments for K-dim c = 0..31 and 32..63
    v16h qf[2];
#pragma unroll
    for (int cc = 0; cc < 2; cc++) {
        const _Float16* qp = Q + (rowOff + sbase + col) * D_ + h * C_ + cc * 32 + hi * 8;
        union { v16h v; v8h p[2]; } u;
        u.p[0] = *(const v8h*)qp;
        u.p[1] = *(const v8h*)(qp + 16);
        qf[cc] = u.v;
    }

    unsigned char rm[8];
#pragma unroll
    for (int r = 0; r < 8; r++) rm[r] = seqMask[rowOff + sbase + r + 8 * hi];

    float rmax[8], rsum[8];
#pragma unroll
    for (int r = 0; r < 8; r++) { rmax[r] = -3.0e38f; rsum[r] = 0.0f; }
    v8f oacc[4];
#pragma unroll
    for (int b = 0; b < 4; b++) oacc[b] = 0.0f;

    const _Float16* Kb  = Kp + rowOff * D_ + h * C_;
    const _Float16* Vtb = Vt + ((size_t)nb * D_ + h * C_) * SEQ;

    for (int t0 = 0; t0 < SEQ; t0 += 32) {
        // two 16x16 score tiles; K rows read as two 16B chunks per lane
        v8f s0 = 0.0f, s1 = 0.0f;
#pragma unroll
        for (int tt = 0; tt < 2; tt++) {
            v8f sc = 0.0f;
#pragma unroll
            for (int cc = 0; cc < 2; cc++) {
                const _Float16* kp = Kb + (size_t)(t0 + tt * 16 + col) * D_ + cc * 32 + hi * 8;
                union { v16h v; v8h p[2]; } ub;
                ub.p[0] = *(const v8h*)kp;
                ub.p[1] = *(const v8h*)(kp + 16);
                v16h bf = ub.v;
                sc = __builtin_amdgcn_wmma_f32_16x16x32_f16(
                    false, qf[cc], false, bf, (short)0, sc, false, false);
            }
            if (tt == 0) s0 = sc; else s1 = sc;
        }

        const unsigned char cm0 = seqMask[rowOff + t0 + col];
        const unsigned char cm1 = seqMask[rowOff + t0 + 16 + col];

        float p0[8], p1[8];
#pragma unroll
        for (int r = 0; r < 8; r++) {
            const int srow = sbase + r + 8 * hi;
            float a0 = s0[r] + attMask[(size_t)srow * SEQ + t0 + col];
            float a1 = s1[r] + attMask[(size_t)srow * SEQ + t0 + 16 + col];
            if (rm[r] && cm0) a0 = -1.0e30f;
            if (rm[r] && cm1) a1 = -1.0e30f;
            float mx = fmaxf(a0, a1);
#pragma unroll
            for (int off = 8; off >= 1; off >>= 1) mx = fmaxf(mx, __shfl_xor(mx, off, 16));
            const float nm = fmaxf(rmax[r], mx);
            const float e0 = __expf(a0 - nm);
            const float e1 = __expf(a1 - nm);
            float ps = e0 + e1;
#pragma unroll
            for (int off = 8; off >= 1; off >>= 1) ps += __shfl_xor(ps, off, 16);
            const float corr = __expf(rmax[r] - nm);
            rsum[r] = rsum[r] * corr + ps;
            rmax[r] = nm;
#pragma unroll
            for (int b = 0; b < 4; b++) oacc[b][r] *= corr;
            p0[r] = e0; p1[r] = e1;
        }

        // stage P (16x32 f16) through LDS to re-shape into A-fragment layout
        __syncthreads();
#pragma unroll
        for (int r = 0; r < 8; r++) {
            const int m = r + 8 * hi;
            lds[m * 32 + col]      = (_Float16)p0[r];
            lds[m * 32 + col + 16] = (_Float16)p1[r];
        }
        __syncthreads();

        v16h pf;
        {
            const _Float16* lp = &lds[col * 32 + hi * 8];
            union { v16h v; v8h p[2]; } u;
            u.p[0] = *(const v8h*)lp;
            u.p[1] = *(const v8h*)(lp + 16);
            pf = u.v;
        }

        // O += P @ V: B-frag from transposed V, two 16B chunks per lane
#pragma unroll
        for (int b = 0; b < 4; b++) {
            const _Float16* vp = Vtb + (size_t)(b * 16 + col) * SEQ + t0 + hi * 8;
            union { v16h v; v8h p[2]; } uv;
            uv.p[0] = *(const v8h*)vp;
            uv.p[1] = *(const v8h*)(vp + 16);
            v16h vf = uv.v;
            oacc[b] = __builtin_amdgcn_wmma_f32_16x16x32_f16(
                false, pf, false, vf, (short)0, oacc[b], false, false);
        }
    }

    // normalize + sigmoid gate, store f16
#pragma unroll
    for (int r = 0; r < 8; r++) {
        const int srow = sbase + r + 8 * hi;
        const float inv = 1.0f / rsum[r];
#pragma unroll
        for (int b = 0; b < 4; b++) {
            const size_t idx = (rowOff + srow) * (size_t)D_ + h * C_ + b * 16 + col;
            O[idx] = (_Float16)(oacc[b][r] * inv * G[idx]);
        }
    }
}

// ---------------------------------------------------------------------------
// Host launcher
// ---------------------------------------------------------------------------
extern "C" void kernel_launch(void* const* d_in, const int* in_sizes, int n_in,
                              void* d_out, int out_size, void* d_ws, size_t ws_size,
                              hipStream_t stream)
{
    (void)in_sizes; (void)n_in; (void)out_size; (void)ws_size;

    const float* Qin = (const float*)d_in[0];
    const float* Kin = (const float*)d_in[1];
    const float* Vin = (const float*)d_in[2];
    const unsigned char* seqMask = (const unsigned char*)d_in[3];
    const float* attMask = (const float*)d_in[4];
    const float* Wq = (const float*)d_in[5];
    const float* Wk = (const float*)d_in[6];
    const float* Wv = (const float*)d_in[7];
    const float* Wg = (const float*)d_in[8];
    const float* bg = (const float*)d_in[9];
    const float* Wo = (const float*)d_in[10];
    const float* bo = (const float*)d_in[11];

    char* ws = (char*)d_ws;
    size_t off = 0;
    auto carve = [&](size_t bytes) { void* p = ws + off; off += bytes; return p; };

    const size_t ACT_E = (size_t)MROWS * D_;   // 2048*1024
    const size_t W_E   = (size_t)D_ * D_;      // 1024*1024

    _Float16* hQin = (_Float16*)carve(ACT_E * 2);
    _Float16* hKin = (_Float16*)carve(ACT_E * 2);
    _Float16* hVin = (_Float16*)carve(ACT_E * 2);
    _Float16* tWq  = (_Float16*)carve(W_E * 2);   // transposed f16 weights [N][K]
    _Float16* tWk  = (_Float16*)carve(W_E * 2);
    _Float16* tWv  = (_Float16*)carve(W_E * 2);
    _Float16* tWg  = (_Float16*)carve(W_E * 2);
    _Float16* tWo  = (_Float16*)carve(W_E * 2);
    _Float16* hQ   = (_Float16*)carve(ACT_E * 2);
    _Float16* hK   = (_Float16*)carve(ACT_E * 2);
    _Float16* hV   = (_Float16*)carve(ACT_E * 2);
    _Float16* hVt  = (_Float16*)carve(ACT_E * 2); // per-head transposed V
    float*    Gbuf = (float*)   carve(ACT_E * 4);
    _Float16* hO   = (_Float16*)carve(ACT_E * 2);

    // 1) fp32 -> f16 conversions (activations) + transpose-convert (weights)
    {
        const int tb = 256;
        cvt_f32_to_f16<<<(ACT_E / 4 + tb - 1) / tb, tb, 0, stream>>>(Qin, hQin, (int)ACT_E);
        cvt_f32_to_f16<<<(ACT_E / 4 + tb - 1) / tb, tb, 0, stream>>>(Kin, hKin, (int)ACT_E);
        cvt_f32_to_f16<<<(ACT_E / 4 + tb - 1) / tb, tb, 0, stream>>>(Vin, hVin, (int)ACT_E);
        dim3 tg(D_ / 32, D_ / 32), tbk(32, 8);
        transpose_cvt<<<tg, tbk, 0, stream>>>(Wq, tWq, D_, D_);
        transpose_cvt<<<tg, tbk, 0, stream>>>(Wk, tWk, D_, D_);
        transpose_cvt<<<tg, tbk, 0, stream>>>(Wv, tWv, D_, D_);
        transpose_cvt<<<tg, tbk, 0, stream>>>(Wg, tWg, D_, D_);
        transpose_cvt<<<tg, tbk, 0, stream>>>(Wo, tWo, D_, D_);
    }

    // 2) projections (WMMA GEMMs).  Q gets the 1/sqrt(C)=0.125 scale folded in.
    dim3 gg(MROWS / 128, D_ / 128), gb(128);
    wmma_gemm<<<gg, gb, 0, stream>>>(hQin, tWq, nullptr, 0.125f, 0, nullptr, hQ, MROWS, D_, D_);
    wmma_gemm<<<gg, gb, 0, stream>>>(hKin, tWk, nullptr, 1.0f,   0, nullptr, hK, MROWS, D_, D_);
    wmma_gemm<<<gg, gb, 0, stream>>>(hVin, tWv, nullptr, 1.0f,   0, nullptr, hV, MROWS, D_, D_);
    wmma_gemm<<<gg, gb, 0, stream>>>(hVin, tWg, bg,      1.0f,   1, Gbuf, nullptr, MROWS, D_, D_);

    // 2b) per-head transpose of V for contiguous P@V fragments
    transpose_V<<<dim3(SEQ / 32, D_ / 32, NB), dim3(32, 8), 0, stream>>>(hV, hVt);

    // 3) flash attention + gate
    attn_kernel<<<dim3(SEQ / 16, H_, NB), dim3(32), 0, stream>>>(
        hQ, hK, hVt, Gbuf, seqMask, attMask, hO);

    // 4) output projection + bias -> fp32 d_out
    wmma_gemm<<<gg, gb, 0, stream>>>(hO, tWo, bo, 1.0f, 0, (float*)d_out, nullptr, MROWS, D_, D_);
}
